// DWM_10514079940641
// MI455X (gfx1250) — compile-verified, matrix operand order checked by reading, not run on hardware
//
#include <hip/hip_runtime.h>
#include <hip/hip_bf16.h>

// ---------------- problem constants (from reference) ----------------
#define BATCH     16
#define SEQ       512
#define IN_DIM    258          // CB + DB
#define ST_DIM    512          // hidden state
#define OUT_DIM   256
#define MEMM      64           // memory word length M
#define NADDR     512          // memory addresses N
#define HEADS     2
#define UPDH      202          // per-head interface params
#define UPDSZ     404          // HEADS*UPDH
#define READSZ    128          // HEADS*MEMM
#define CTRL_K    898          // IN_DIM + ST_DIM + READSZ
#define NTOT      1172         // ST_DIM + OUT_DIM + UPDSZ
#define KT        29           // ceil(898/32)  K tiles (K padded to 928)
#define NTP       80           // N tiles padded to 8 waves * 10 (74 real + 6 zero)
#define KPAD      928
#define NTW       10           // N tiles per wave, exact (no guards)
#define NTC       5            // N tiles per load/compute chunk (reg pressure)
#define EPSR      1e-12f

typedef __attribute__((ext_vector_type(16))) _Float16 v16h;
typedef __attribute__((ext_vector_type(8)))  float    v8f;

// per-lane K index inside a 32-wide K tile for 16-bit WMMA A/B fragments
// (ISA 7.12.2: VGPR0..3 hold K pairs 0..7 (lanes<16) / 8..15 (lanes>=16),
//  VGPR4..7 hold K 16..23 / 24..31)
__device__ __forceinline__ int kidx16(int lane, int j) {
    int v = j >> 1, p = j & 1;
    int k = (v < 4) ? (v * 2 + p) : (16 + (v - 4) * 2 + p);
    if (lane & 16) k += 8;
    return k;
}

__device__ __forceinline__ float sigm(float x) { return 1.f / (1.f + __expf(-x)); }
__device__ __forceinline__ float softp(float x) {
    return (x > 20.f) ? x : __logf(1.f + __expf(x));
}

// ---------------- LDS layout (float offsets) ----------------
#define L_MEM     0            // 64*512
#define L_STATE   32768        // 512
#define L_WT      33280        // 2*512
#define L_WD      34304        // 2*512
#define L_WCUR    35328        // 2*512
#define L_Y       36352        // 1280 (80 N-tiles; reused as wt_k later)
#define L_NORM    37632        // 512
#define L_READ    38144        // 128
#define L_PAR     38272        // 416 (404 used)
#define L_KN      38688        // 128
#define L_EA      38816        // 128
#define L_AA      38944        // 128
#define L_SC      39072        // 32 scalars: [h*16 + i]
#define L_RED     39104        // 16 reduction scratch
#define L_COMBH   39120        // 928 halves = 464 floats
#define LDS_FLOATS 39584
#define LDS_BYTES  (LDS_FLOATS * 4)

__device__ __forceinline__ float blockReduceSum(float v, float* red) {
#pragma unroll
    for (int o = 16; o > 0; o >>= 1) v += __shfl_xor(v, o, 32);
    int w = threadIdx.x >> 5;
    __syncthreads();
    if ((threadIdx.x & 31) == 0) red[w] = v;
    __syncthreads();
    float s = 0.f;
#pragma unroll
    for (int i = 0; i < 8; i++) s += red[i];
    return s;
}

__device__ __forceinline__ float blockReduceMax(float v, float* red) {
#pragma unroll
    for (int o = 16; o > 0; o >>= 1) v = fmaxf(v, __shfl_xor(v, o, 32));
    int w = threadIdx.x >> 5;
    __syncthreads();
    if ((threadIdx.x & 31) == 0) red[w] = v;
    __syncthreads();
    float s = -3.4e38f;
#pragma unroll
    for (int i = 0; i < 8; i++) s = fmaxf(s, red[i]);
    return s;
}

// ---------------- prep: pack fused fp16 weights into WMMA B-fragment tiles ----
// tile(kt,nt): 32 lanes x v16h (512 halves, 1KB). element (lane,j) = W[kt*32 +
// kidx16(lane,j)][nt*16 + (lane&15)] with fused columns [W_s | W_o | W_u].
// Tiles with nt >= 74 or k >= 898 are zero padding (keeps GEMM loop guard-free).
__global__ void dwm_prep(const float* __restrict__ Ws, const float* __restrict__ Wo,
                         const float* __restrict__ Wu, _Float16* __restrict__ Wp) {
    int tile = blockIdx.x;           // 0 .. KT*NTP-1
    int kt = tile / NTP, nt = tile % NTP;
    int t = threadIdx.x;             // 0..511
    int lane = t >> 4, j = t & 15;
    int k = kt * 32 + kidx16(lane, j);
    int n = nt * 16 + (lane & 15);
    float v = 0.f;
    if (k < CTRL_K) {
        if (n < ST_DIM)                 v = Ws[k * ST_DIM + n];
        else if (n < ST_DIM + OUT_DIM)  v = Wo[k * OUT_DIM + (n - ST_DIM)];
        else if (n < NTOT)              v = Wu[k * UPDSZ + (n - ST_DIM - OUT_DIM)];
    }
    Wp[(size_t)tile * 512 + lane * 16 + j] = (_Float16)v;
}

// ---------------- main: one persistent workgroup (8 wave32) per batch --------
// 8 waves = exactly 2 waves/SIMD; LDS use caps the WGP at 1 workgroup, so let
// the register allocator budget the VGPR file for 2 waves/EU (no spilling).
__global__ void __launch_bounds__(256)
__attribute__((amdgpu_waves_per_eu(2)))
dwm_main(const float* __restrict__ x, const float* __restrict__ bs,
         const float* __restrict__ bo, const float* __restrict__ bu,
         const _Float16* __restrict__ Wp, float* __restrict__ out) {
    extern __shared__ float lds[];
    const int tid  = threadIdx.x;            // 0..255
    const int lane = tid & 31;
    const int wave = tid >> 5;               // 0..7
    const int b    = blockIdx.x;             // batch element

    float* mem    = lds + L_MEM;
    float* state  = lds + L_STATE;
    float* wt     = lds + L_WT;
    float* wdyn   = lds + L_WD;
    float* wcur   = lds + L_WCUR;
    float* y      = lds + L_Y;
    float* cnorm  = lds + L_NORM;
    float* rdv    = lds + L_READ;
    float* par    = lds + L_PAR;
    float* kn     = lds + L_KN;
    float* eA     = lds + L_EA;
    float* aA     = lds + L_AA;
    float* sc     = lds + L_SC;
    float* red    = lds + L_RED;
    _Float16* combh = (_Float16*)(lds + L_COMBH);
    const v16h* Wv = (const v16h*)Wp;

    // ---- init carry ----
    for (int i = tid; i < MEMM * NADDR; i += 256) mem[i] = 0.01f;
    for (int i = tid; i < ST_DIM; i += 256) state[i] = 1.f;
    for (int i = tid; i < HEADS * NADDR; i += 256) {
        float v = ((i & (NADDR - 1)) == 0) ? 1.f : 0.f;
        wt[i] = v; wdyn[i] = v;
    }
    __syncthreads();

    for (int t = 0; t < SEQ; ++t) {
        // ---- (a) read = einsum(wt, mem) : 128 outputs ----
        if (tid < READSZ) {
            int h = tid >> 6, m = tid & 63;
            const float* wrow = wt + h * NADDR;
            const float* mrow = mem + m * NADDR;
            float s = 0.f;
            for (int n = 0; n < NADDR; ++n) s += wrow[n] * mrow[n];
            rdv[tid] = s;
        }
        __syncthreads();

        // ---- (b) comb -> fp16 (padded to 928) ----
        const float* xt = x + ((size_t)b * SEQ + t) * IN_DIM;
        for (int i = tid; i < KPAD; i += 256) {
            float v;
            if (i < IN_DIM)                  v = xt[i];
            else if (i < IN_DIM + ST_DIM)    v = state[i - IN_DIM];
            else if (i < CTRL_K)             v = rdv[i - IN_DIM - ST_DIM];
            else                             v = 0.f;
            combh[i] = (_Float16)v;
        }
        __syncthreads();

        // ---- (c) GEMV via WMMA: y = comb @ [W_s|W_o|W_u], guard-free ----
        // 10 N-tile accumulators per wave; B fragments streamed in 2 chunks of
        // 5 (clause of 10 b128 loads -> 5 back-to-back WMMAs) to stay within
        // the 2-wave/SIMD register budget (no scratch spills).
        {
            v8f acc[NTW];
#pragma unroll
            for (int i = 0; i < NTW; ++i) acc[i] = (v8f){};
            const v16h* wvw = Wv + (size_t)wave * 32;   // this wave's first tile
            for (int kt = 0; kt < KT; ++kt) {
                union { v16h v; _Float16 h[16]; } a;
#pragma unroll
                for (int j = 0; j < 16; ++j)
                    a.h[j] = combh[kt * 32 + kidx16(lane, j)];
#pragma unroll
                for (int c = 0; c < NTW / NTC; ++c) {
                    v16h bf[NTC];
#pragma unroll
                    for (int i = 0; i < NTC; ++i)
                        bf[i] = wvw[((size_t)kt * NTP + (c * NTC + i) * 8) * 32 +
                                    lane];
#pragma unroll
                    for (int i = 0; i < NTC; ++i)
                        acc[c * NTC + i] = __builtin_amdgcn_wmma_f32_16x16x32_f16(
                            false, a.v, false, bf[i], (short)0, acc[c * NTC + i],
                            false, false);
                }
            }
#pragma unroll
            for (int i = 0; i < NTW; ++i) {
                int nt = wave + i * 8;
                if (lane < 16) y[nt * 16 + lane] = acc[i][0];
            }
        }
        __syncthreads();

        // ---- (d) state / out / interface params ----
        for (int i = tid; i < ST_DIM; i += 256) state[i] = sigm(y[i] + bs[i]);
        if (tid < OUT_DIM)
            out[((size_t)b * SEQ + t) * OUT_DIM + tid] =
                sigm(y[ST_DIM + tid] + bo[tid]);
        for (int i = tid; i < UPDSZ; i += 256)
            par[i] = y[ST_DIM + OUT_DIM + i] + bu[i];
        __syncthreads();

        // ---- (e) per-head scalars + key norm + erase/add prep ----
        if (tid < HEADS) {
            int h = tid; const float* p = par + h * UPDH; float* s = sc + h * 16;
            float s0 = softp(p[0]), s1 = softp(p[1]), s2 = softp(p[2]);
            float mx = fmaxf(s0, fmaxf(s1, s2));
            float e0 = __expf(s0 - mx), e1 = __expf(s1 - mx), e2 = __expf(s2 - mx);
            float es = e0 + e1 + e2;
            s[0] = e0 / es; s[1] = e1 / es; s[2] = e2 / es;
            s[3] = sigm(p[3]);                                   // jd
            float j0 = p[4], j1 = p[5], j2 = p[6];
            mx = fmaxf(j0, fmaxf(j1, j2));
            e0 = __expf(j0 - mx); e1 = __expf(j1 - mx); e2 = __expf(j2 - mx);
            es = e0 + e1 + e2;
            s[4] = e0 / es; s[5] = e1 / es; s[6] = e2 / es;      // jump
            s[7] = 1.f + softp(p[7]);                            // gamma
            s[8] = softp(p[200]);                                // beta
            s[9] = sigm(p[201]);                                 // gate
            float ss = 0.f;
            for (int m = 0; m < MEMM; ++m) { float kv = p[136 + m]; ss += kv * kv; }
            s[10] = 1.f / (sqrtf(ss) + EPSR);                    // 1/(|k|+eps)
        }
        // memory column norms (before mem update)
        {
            int n0 = tid, n1 = tid + 256;
            float a = 0.f, c = 0.f;
            for (int m = 0; m < MEMM; ++m) {
                float v0 = mem[m * NADDR + n0]; a += v0 * v0;
                float v1 = mem[m * NADDR + n1]; c += v1 * v1;
            }
            cnorm[n0] = sqrtf(a); cnorm[n1] = sqrtf(c);
        }
        __syncthreads();
        if (tid < HEADS * MEMM) {
            int h = tid >> 6, m = tid & 63;
            kn[tid] = par[h * UPDH + 136 + m] * sc[h * 16 + 10];
            eA[tid] = sigm(par[h * UPDH + 8 + m]);
            aA[tid] = par[h * UPDH + 72 + m];
        }
        __syncthreads();

        // ---- (f) content addressing: wt_k = softmax(beta * cos(k, mem)) ----
        for (int h = 0; h < HEADS; ++h) {
            float beta = sc[h * 16 + 8];
            int n0 = tid, n1 = tid + 256;
            float d0 = 0.f, d1 = 0.f;
            const float* kh = kn + h * MEMM;
            for (int m = 0; m < MEMM; ++m) {
                d0 += kh[m] * mem[m * NADDR + n0];
                d1 += kh[m] * mem[m * NADDR + n1];
            }
            float v0 = beta * d0 / (cnorm[n0] + EPSR);
            float v1 = beta * d1 / (cnorm[n1] + EPSR);
            float mx = blockReduceMax(fmaxf(v0, v1), red);
            float e0 = __expf(v0 - mx), e1 = __expf(v1 - mx);
            float es = blockReduceSum(e0 + e1, red);
            y[h * NADDR + n0] = e0 / es;      // y reused as wt_k
            y[h * NADDR + n1] = e1 / es;
        }
        __syncthreads();

        // ---- (g) bookmark update, jump mix, gate ----
        for (int i = tid; i < HEADS * NADDR; i += 256) {
            int h = i >> 9, n = i & (NADDR - 1);
            const float* s = sc + h * 16;
            float jd = s[3];
            float wd = (1.f - jd) * wdyn[i] + jd * wt[i];
            wdyn[i] = wd;
            float wfix = (n == 0) ? 1.f : 0.f;
            float wj = s[4] * wt[i] + s[5] * wd + s[6] * wfix;
            float g = s[9];
            wcur[i] = g * y[i] + (1.f - g) * wj;
        }
        __syncthreads();

        // ---- (h) circular shift conv + sharpen ----
        for (int h = 0; h < HEADS; ++h) {
            const float* s = sc + h * 16;
            float gamma = s[7];
            const float* wc = wcur + h * NADDR;
            int n0 = tid, n1 = tid + 256;
            float w0 = s[0] * wc[(n0 + 1) & (NADDR - 1)] + s[1] * wc[n0] +
                       s[2] * wc[(n0 + NADDR - 1) & (NADDR - 1)];
            float w1 = s[0] * wc[(n1 + 1) & (NADDR - 1)] + s[1] * wc[n1] +
                       s[2] * wc[(n1 + NADDR - 1) & (NADDR - 1)];
            float p0 = (w0 > 0.f) ? __expf(gamma * __logf(w0)) : 0.f;
            float p1 = (w1 > 0.f) ? __expf(gamma * __logf(w1)) : 0.f;
            float ps = blockReduceSum(p0 + p1, red);
            float inv = 1.f / (ps + EPSR);
            wt[h * NADDR + n0] = p0 * inv;
            wt[h * NADDR + n1] = p1 * inv;
        }
        __syncthreads();

        // ---- (i) memory erase + add ----
        for (int e = tid; e < MEMM * NADDR; e += 256) {
            int m = e >> 9, n = e & (NADDR - 1);
            float w0 = wt[n], w1 = wt[NADDR + n];
            float er = eA[m] * w0 + eA[MEMM + m] * w1;
            float ad = aA[m] * w0 + aA[MEMM + m] * w1;
            mem[e] = mem[e] * (1.f - er) + ad;
        }
        __syncthreads();
    }
}

extern "C" void kernel_launch(void* const* d_in, const int* in_sizes, int n_in,
                              void* d_out, int out_size, void* d_ws, size_t ws_size,
                              hipStream_t stream) {
    const float* inputs = (const float*)d_in[0];
    const float* Ws     = (const float*)d_in[1];
    const float* bs     = (const float*)d_in[2];
    const float* Wo     = (const float*)d_in[3];
    const float* bo     = (const float*)d_in[4];
    const float* Wu     = (const float*)d_in[5];
    const float* bu     = (const float*)d_in[6];
    float* out          = (float*)d_out;
    _Float16* Wp        = (_Float16*)d_ws;   // needs KT*NTP*512*2 = ~2.4 MB

    // pack fused fp16 weights in WMMA fragment layout (L2-resident thereafter)
    dwm_prep<<<dim3(KT * NTP), dim3(512), 0, stream>>>(Ws, Wo, Wu, Wp);
    // one persistent workgroup per batch element; full carry lives in LDS
    dwm_main<<<dim3(BATCH), dim3(256), LDS_BYTES, stream>>>(
        inputs, bs, bo, bu, Wp, out);
}
